// Transformer_73753178407459
// MI455X (gfx1250) — matrix operand order repaired and checked by reading, not compiled, not run on hardware
//
#include <hip/hip_runtime.h>

typedef __bf16 bf16;
typedef __attribute__((ext_vector_type(8)))  bf16  v8bf;
typedef __attribute__((ext_vector_type(16))) bf16  v16bf;
typedef __attribute__((ext_vector_type(8)))  float v8f;

#define DIMX 768
#define NTOK 8192   // B*N
#define SEQ  1024
#define NHEADS 12
#define DH   64
#define QKVN 2304
#define FFN  3072

static __device__ inline v16bf mk16(v8bf lo, v8bf hi) {
  v16bf r;
#pragma unroll
  for (int i = 0; i < 8; i++) { r[i] = lo[i]; r[i + 8] = hi[i]; }
  return r;
}

// ---------------- LayerNorm (fp32 in) -> bf16 out ----------------
__global__ __launch_bounds__(256)
void ln_kernel(const float* __restrict__ x, const float* __restrict__ g,
               const float* __restrict__ beta, bf16* __restrict__ out) {
  __shared__ float ssum[256], ssq[256];
  int row = blockIdx.x, t = threadIdx.x;
  const float* xr = x + (size_t)row * DIMX;
  float a0 = xr[t], a1 = xr[t + 256], a2 = xr[t + 512];
  ssum[t] = a0 + a1 + a2;
  ssq[t]  = a0 * a0 + a1 * a1 + a2 * a2;
  __syncthreads();
  for (int s = 128; s > 0; s >>= 1) {
    if (t < s) { ssum[t] += ssum[t + s]; ssq[t] += ssq[t + s]; }
    __syncthreads();
  }
  float mu  = ssum[0] * (1.0f / DIMX);
  float var = ssq[0] * (1.0f / DIMX) - mu * mu;
  float rs  = rsqrtf(var + 1e-5f);
  bf16* orow = out + (size_t)row * DIMX;
  orow[t]       = (bf16)((a0 - mu) * rs * g[t]       + beta[t]);
  orow[t + 256] = (bf16)((a1 - mu) * rs * g[t + 256] + beta[t + 256]);
  orow[t + 512] = (bf16)((a2 - mu) * rs * g[t + 512] + beta[t + 512]);
}

// ---------------- Weight transpose + fp32 -> bf16: W[K][N] -> WT[N][K] ----------------
__global__ __launch_bounds__(256)
void transpose_kernel(const float* __restrict__ W, bf16* __restrict__ WT, int K, int N) {
  __shared__ float tile[32][33];
  int n0 = blockIdx.x * 32, k0 = blockIdx.y * 32;
  int tx = threadIdx.x, ty = threadIdx.y;
#pragma unroll
  for (int i = 0; i < 4; i++)
    tile[ty + i * 8][tx] = W[(size_t)(k0 + ty + i * 8) * N + n0 + tx];
  __syncthreads();
#pragma unroll
  for (int i = 0; i < 4; i++)
    WT[(size_t)(n0 + ty + i * 8) * K + k0 + tx] = (bf16)tile[tx][ty + i * 8];
}

// ---------------- bf16 WMMA GEMM, double-buffered: out = A[MxK] @ BT[NxK]^T + bias ----
// MODE 0: store bf16; MODE 1: gelu -> bf16; MODE 2: accumulate into fp32 residual
template <int MODE>
__global__ __launch_bounds__(256)
void gemm_kernel(const bf16* __restrict__ A, const bf16* __restrict__ BT,
                 const float* __restrict__ bias, bf16* __restrict__ outb,
                 float* __restrict__ outf, int M, int N, int K) {
  __shared__ __align__(16) bf16 As[2][128 * 32];
  __shared__ __align__(16) bf16 Bs[2][128 * 32];
  int t = threadIdx.x, lane = t & 31, wid = t >> 5;
  int wm = wid >> 1, wn = wid & 1;
  int lr = lane & 15, lg = lane >> 4;
  int m0 = blockIdx.y * 128, n0 = blockIdx.x * 128;
  // this thread's two staging slots: rows/segments of the 128x32 tile
  int r0 = (t * 2) >> 2, s0 = (t * 2) & 3;
  int r1 = (t * 2 + 1) >> 2, s1 = (t * 2 + 1) & 3;

  v8f acc[2][4];
#pragma unroll
  for (int i = 0; i < 2; i++)
#pragma unroll
    for (int j = 0; j < 4; j++) acc[i][j] = (v8f)(0.0f);

  // preload tile 0 straight into LDS buffer 0
  *(v8bf*)&As[0][r0 * 32 + s0 * 8] = *(const v8bf*)&A[(size_t)(m0 + r0) * K + s0 * 8];
  *(v8bf*)&As[0][r1 * 32 + s1 * 8] = *(const v8bf*)&A[(size_t)(m0 + r1) * K + s1 * 8];
  *(v8bf*)&Bs[0][r0 * 32 + s0 * 8] = *(const v8bf*)&BT[(size_t)(n0 + r0) * K + s0 * 8];
  *(v8bf*)&Bs[0][r1 * 32 + s1 * 8] = *(const v8bf*)&BT[(size_t)(n0 + r1) * K + s1 * 8];

  int nk = K / 32;
  for (int kt = 0; kt < nk; kt++) {
    int cur = kt & 1;
    bool pre = (kt + 1) < nk;
    v8bf sa0, sa1, sb0, sb1;
    if (pre) {  // issue next tile's global loads before compute
      int kn = (kt + 1) * 32;
      sa0 = *(const v8bf*)&A[(size_t)(m0 + r0) * K + kn + s0 * 8];
      sa1 = *(const v8bf*)&A[(size_t)(m0 + r1) * K + kn + s1 * 8];
      sb0 = *(const v8bf*)&BT[(size_t)(n0 + r0) * K + kn + s0 * 8];
      sb1 = *(const v8bf*)&BT[(size_t)(n0 + r1) * K + kn + s1 * 8];
    }
    __syncthreads();  // LDS buffer `cur` ready; prior reads of `cur^1` done

    const bf16* Ac = As[cur];
    const bf16* Bc = Bs[cur];
    v16bf af[2], bfrag[4];
#pragma unroll
    for (int mi = 0; mi < 2; mi++) {
      int r = wm * 32 + mi * 16 + lr;
      af[mi] = mk16(*(const v8bf*)&Ac[r * 32 + lg * 8], *(const v8bf*)&Ac[r * 32 + 16 + lg * 8]);
    }
#pragma unroll
    for (int ni = 0; ni < 4; ni++) {
      int r = wn * 64 + ni * 16 + lr;
      bfrag[ni] = mk16(*(const v8bf*)&Bc[r * 32 + lg * 8], *(const v8bf*)&Bc[r * 32 + 16 + lg * 8]);
    }
#pragma unroll
    for (int mi = 0; mi < 2; mi++)
#pragma unroll
      for (int ni = 0; ni < 4; ni++)
        acc[mi][ni] = __builtin_amdgcn_wmma_f32_16x16x32_bf16(
            false, af[mi], false, bfrag[ni], (short)0, acc[mi][ni], false, false);

    if (pre) {  // park the staged tile in the other LDS buffer (waits happen post-compute)
      int nxt = cur ^ 1;
      *(v8bf*)&As[nxt][r0 * 32 + s0 * 8] = sa0;
      *(v8bf*)&As[nxt][r1 * 32 + s1 * 8] = sa1;
      *(v8bf*)&Bs[nxt][r0 * 32 + s0 * 8] = sb0;
      *(v8bf*)&Bs[nxt][r1 * 32 + s1 * 8] = sb1;
    }
  }

#pragma unroll
  for (int ni = 0; ni < 4; ni++) {
    int col = n0 + wn * 64 + ni * 16 + lr;
    float bv = bias[col];
#pragma unroll
    for (int mi = 0; mi < 2; mi++) {
#pragma unroll
      for (int r = 0; r < 8; r++) {
        int row = m0 + wm * 32 + mi * 16 + r + lg * 8;
        float v = acc[mi][ni][r] + bv;
        if (MODE == 1) v = 0.5f * v * (1.0f + erff(v * 0.70710678118654752f));
        if (MODE == 2) outf[(size_t)row * N + col] += v;
        else           outb[(size_t)row * N + col] = (bf16)v;
      }
    }
  }
}

// ---------------- Flash attention, one block per (b, h, 128-query tile) ----------------
__global__ __launch_bounds__(256)
void attn_kernel(const bf16* __restrict__ qkv, float* __restrict__ x) {
  __shared__ __align__(16) bf16 Ks[32 * 64];
  __shared__ __align__(16) bf16 Vt[64 * 32];
  __shared__ __align__(16) bf16 Ps[8][16 * 32];
  int t = threadIdx.x, lane = t & 31, wid = t >> 5;
  int lr = lane & 15, lg = lane >> 4;
  int b = blockIdx.z, h = blockIdx.y, qb = blockIdx.x;
  size_t tokbase = (size_t)b * SEQ;
  int qoff = h * DH, koff = DIMX + h * DH, voff = 2 * DIMX + h * DH;

  v16bf qf[2];
  {
    size_t row = tokbase + qb * 128 + wid * 16 + lr;
    const bf16* qr = qkv + row * QKVN + qoff;
#pragma unroll
    for (int ks = 0; ks < 2; ks++)
      qf[ks] = mk16(*(const v8bf*)&qr[ks * 32 + lg * 8],
                    *(const v8bf*)&qr[ks * 32 + 16 + lg * 8]);
  }
  v16bf ones;
#pragma unroll
  for (int i = 0; i < 16; i++) ones[i] = (bf16)1.0f;

  float mrow[8];
  v8f o[4], lsum;
#pragma unroll
  for (int r = 0; r < 8; r++) mrow[r] = -1e30f;
#pragma unroll
  for (int d = 0; d < 4; d++) o[d] = (v8f)(0.0f);
  lsum = (v8f)(0.0f);
  const float scale = 0.03608439182435161f;  // 1/sqrt(768): module scales by full dim

  for (int kb = 0; kb < SEQ / 32; kb++) {
    __syncthreads();
    {
      int m = t >> 3, seg = t & 7;
      size_t krow = tokbase + kb * 32 + m;
      *(v8bf*)&Ks[m * 64 + seg * 8] = *(const v8bf*)&qkv[krow * QKVN + koff + seg * 8];
      v8bf vv = *(const v8bf*)&qkv[krow * QKVN + voff + seg * 8];
#pragma unroll
      for (int j = 0; j < 8; j++) Vt[(seg * 8 + j) * 32 + m] = vv[j];
    }
    __syncthreads();

    // S = Q @ K^T  (16 q-rows x 32 keys per wave, K-dim = dh = 64)
    v8f s[2];
#pragma unroll
    for (int c = 0; c < 2; c++) {
      s[c] = (v8f)(0.0f);
#pragma unroll
      for (int ds = 0; ds < 2; ds++) {
        const bf16* kr = &Ks[(c * 16 + lr) * 64 + ds * 32 + lg * 8];
        v16bf kf = mk16(*(const v8bf*)kr, *(const v8bf*)(kr + 16));
        s[c] = __builtin_amdgcn_wmma_f32_16x16x32_bf16(
            false, qf[ds], false, kf, (short)0, s[c], false, false);
      }
    }

    // Online softmax: max via lane-shuffle; row-sum via WMMA against all-ones B
#pragma unroll
    for (int r = 0; r < 8; r++) {
      float s0 = s[0][r] * scale, s1 = s[1][r] * scale;
      float rm = fmaxf(s0, s1);
#pragma unroll
      for (int msk = 8; msk >= 1; msk >>= 1) rm = fmaxf(rm, __shfl_xor(rm, msk, 32));
      float mn = fmaxf(mrow[r], rm);
      float corr = __expf(mrow[r] - mn);
      mrow[r] = mn;
      float p0 = __expf(s0 - mn), p1 = __expf(s1 - mn);
#pragma unroll
      for (int d = 0; d < 4; d++) o[d][r] *= corr;
      lsum[r] *= corr;
      int prow = r + lg * 8;
      Ps[wid][prow * 32 + lr]      = (bf16)p0;
      Ps[wid][prow * 32 + 16 + lr] = (bf16)p1;
    }

    // A-frag of P from per-wave LDS region
    v16bf pf = mk16(*(v8bf*)&Ps[wid][lr * 32 + lg * 8],
                    *(v8bf*)&Ps[wid][lr * 32 + 16 + lg * 8]);
    // lsum += P @ ones  (row sums land replicated across the row's lanes)
    lsum = __builtin_amdgcn_wmma_f32_16x16x32_bf16(
        false, pf, false, ones, (short)0, lsum, false, false);
    // O += P @ V
#pragma unroll
    for (int d = 0; d < 4; d++) {
      const bf16* vr = &Vt[(d * 16 + lr) * 32 + lg * 8];
      v16bf vf = mk16(*(const v8bf*)vr, *(const v8bf*)(vr + 16));
      o[d] = __builtin_amdgcn_wmma_f32_16x16x32_bf16(
          false, pf, false, vf, (short)0, o[d], false, false);
    }
  }

  // residual add: x += softmax(QK^T) V
#pragma unroll
  for (int d = 0; d < 4; d++) {
#pragma unroll
    for (int r = 0; r < 8; r++) {
      size_t row = tokbase + qb * 128 + wid * 16 + r + lg * 8;
      int col = h * DH + d * 16 + lr;
      x[row * DIMX + col] += o[d][r] / lsum[r];
    }
  }
}

extern "C" void kernel_launch(void* const* d_in, const int* in_sizes, int n_in,
                              void* d_out, int out_size, void* d_ws, size_t ws_size,
                              hipStream_t stream) {
  (void)in_sizes; (void)n_in; (void)out_size; (void)ws_size;
  const float* in_x  = (const float*)d_in[0];
  const float* ln1_g = (const float*)d_in[1];
  const float* ln1_b = (const float*)d_in[2];
  const float* wqkv  = (const float*)d_in[3];
  const float* bqkv  = (const float*)d_in[4];
  const float* ln2_g = (const float*)d_in[5];
  const float* ln2_b = (const float*)d_in[6];
  const float* w1    = (const float*)d_in[7];
  const float* b1    = (const float*)d_in[8];
  const float* w2    = (const float*)d_in[9];
  const float* b2    = (const float*)d_in[10];

  char* ws = (char*)d_ws;
  float* x  = (float*)ws; ws += (size_t)NTOK * DIMX * 4;   // fp32 residual stream
  bf16* x1  = (bf16*)ws;  ws += (size_t)NTOK * DIMX * 2;   // LN output (bf16)
  bf16* act = (bf16*)ws;  ws += (size_t)NTOK * FFN  * 2;   // qkv / ffn-hidden (bf16)
  bf16* wT  = (bf16*)ws;                                   // transposed weight (bf16)

  hipMemcpyAsync(x, in_x, (size_t)NTOK * DIMX * 4, hipMemcpyDeviceToDevice, stream);

  for (int l = 0; l < 6; l++) {
    ln_kernel<<<NTOK, 256, 0, stream>>>(x, ln1_g + l * DIMX, ln1_b + l * DIMX, x1);
    transpose_kernel<<<dim3(QKVN / 32, DIMX / 32), dim3(32, 8), 0, stream>>>(
        wqkv + (size_t)l * DIMX * QKVN, wT, DIMX, QKVN);
    gemm_kernel<0><<<dim3(QKVN / 128, NTOK / 128), 256, 0, stream>>>(
        x1, wT, bqkv + (size_t)l * QKVN, act, nullptr, NTOK, QKVN, DIMX);
    attn_kernel<<<dim3(SEQ / 128, NHEADS, 8), 256, 0, stream>>>(act, x);

    ln_kernel<<<NTOK, 256, 0, stream>>>(x, ln2_g + l * DIMX, ln2_b + l * DIMX, x1);
    transpose_kernel<<<dim3(FFN / 32, DIMX / 32), dim3(32, 8), 0, stream>>>(
        w1 + (size_t)l * DIMX * FFN, wT, DIMX, FFN);
    gemm_kernel<1><<<dim3(FFN / 128, NTOK / 128), 256, 0, stream>>>(
        x1, wT, b1 + (size_t)l * FFN, act, nullptr, NTOK, FFN, DIMX);
    transpose_kernel<<<dim3(DIMX / 32, FFN / 32), dim3(32, 8), 0, stream>>>(
        w2 + (size_t)l * FFN * DIMX, wT, FFN, DIMX);
    gemm_kernel<2><<<dim3(DIMX / 128, NTOK / 128), 256, 0, stream>>>(
        act, wT, b2 + (size_t)l * DIMX, nullptr, x, NTOK, DIMX, FFN);
  }

  hipMemcpyAsync(d_out, x, (size_t)NTOK * DIMX * 4, hipMemcpyDeviceToDevice, stream);
}
